// GAT_53412213293758
// MI455X (gfx1250) — matrix operand order repaired and checked by reading, not compiled, not run on hardware
//
#include <hip/hip_runtime.h>

// ---------------------------------------------------------------------------
// GAT on MI455X (gfx1250): bf16 WMMA projections (16x64 register-blocked)
// + L2-resident edge scatter softmax/aggregation
// ---------------------------------------------------------------------------

#define N_NODES 10000
#define N_EDGES 160000
#define F1      512
#define HID     128
#define HEADS   8
#define COLS    (HEADS * HID)   // 1024

typedef __attribute__((ext_vector_type(16))) __bf16 v16bf;
typedef __attribute__((ext_vector_type(8)))  float  v8f;

// ---- helpers --------------------------------------------------------------

__device__ __forceinline__ unsigned short gat_f2bf(float f) {
    unsigned u = __float_as_uint(f);
    unsigned r = u + 0x7FFFu + ((u >> 16) & 1u);   // round-to-nearest-even
    return (unsigned short)(r >> 16);
}

// monotone float <-> uint key for atomicMax-based segment max
__device__ __forceinline__ unsigned gat_fkey(float f) {
    unsigned b = __float_as_uint(f);
    return (b & 0x80000000u) ? ~b : (b | 0x80000000u);
}
__device__ __forceinline__ float gat_funkey(unsigned k) {
    unsigned b = (k & 0x80000000u) ? (k & 0x7FFFFFFFu) : ~k;
    return __uint_as_float(b);
}

// ---- f32 -> bf16 conversion ----------------------------------------------

__global__ void gat_cvt_bf16(const float* __restrict__ in,
                             unsigned short* __restrict__ out, int n) {
    int t = blockIdx.x * blockDim.x + threadIdx.x;
    if (t < n) out[t] = gat_f2bf(in[t]);
}

// ---- zero fill ------------------------------------------------------------

__global__ void gat_zero4(float4* __restrict__ p, int n4) {
    int t = blockIdx.x * blockDim.x + threadIdx.x;
    if (t < n4) p[t] = make_float4(0.f, 0.f, 0.f, 0.f);
}

// ---- projection GEMM: H[n, h*128+d] = sum_f A[n,f] * W[h,d,f] -------------
// A: [Nnodes, F] bf16 row-major; B: [COLS, F] bf16 (W flattened, f contig)
// One wave per 16x64 output tile: one shared A fragment drives 4 WMMAs
// (v_wmma_f32_16x16x32_bf16), amortizing A loads 4x.

__global__ __launch_bounds__(128) void gat_gemm_wmma(
    const __bf16* __restrict__ A, const __bf16* __restrict__ B,
    float* __restrict__ H, int F, int tiles_total) {
    const int wave = threadIdx.x >> 5;
    const int lane = threadIdx.x & 31;
    const int tile = blockIdx.x * 4 + wave;          // wave-uniform
    if (tile >= tiles_total) return;
    const int mtile  = tile >> 4;                    // / (COLS/64)
    const int ngroup = tile & 15;                    // which 64-col group
    const int row  = mtile * 16 + (lane & 15);
    const int col0 = ngroup * 64 + (lane & 15);
    // A frag (16x32 bf16): lane<16 -> K 0..15 ; lane>=16 -> K 16..31 (contig)
    const int aoff = (lane & 16) ? 16 : 0;
    // B frag (32x16 bf16): lane<16 -> K 0..15 ; lane>=16 -> K 16..31 (contig)
    const int boff = (lane & 16) ? 16 : 0;
    const __bf16* arow = A + (size_t)row * F + aoff;
    const __bf16* b0   = B + (size_t)col0 * F + boff;
    const __bf16* b1   = b0 + (size_t)16 * F;
    const __bf16* b2   = b0 + (size_t)32 * F;
    const __bf16* b3   = b0 + (size_t)48 * F;

    v8f c0 = {}, c1 = {}, c2 = {}, c3 = {};
    for (int kk = 0; kk < F; kk += 32) {
        union { v16bf v; uint4 q[2]; } af, bf0, bf1, bf2, bf3;
        af.q[0]  = *(const uint4*)(arow + kk);
        af.q[1]  = *(const uint4*)(arow + kk + 8);
        bf0.q[0] = *(const uint4*)(b0 + kk);
        bf0.q[1] = *(const uint4*)(b0 + kk + 8);
        bf1.q[0] = *(const uint4*)(b1 + kk);
        bf1.q[1] = *(const uint4*)(b1 + kk + 8);
        bf2.q[0] = *(const uint4*)(b2 + kk);
        bf2.q[1] = *(const uint4*)(b2 + kk + 8);
        bf3.q[0] = *(const uint4*)(b3 + kk);
        bf3.q[1] = *(const uint4*)(b3 + kk + 8);
        c0 = __builtin_amdgcn_wmma_f32_16x16x32_bf16(
                false, af.v, false, bf0.v, (short)0, c0, false, false);
        c1 = __builtin_amdgcn_wmma_f32_16x16x32_bf16(
                false, af.v, false, bf1.v, (short)0, c1, false, false);
        c2 = __builtin_amdgcn_wmma_f32_16x16x32_bf16(
                false, af.v, false, bf2.v, (short)0, c2, false, false);
        c3 = __builtin_amdgcn_wmma_f32_16x16x32_bf16(
                false, af.v, false, bf3.v, (short)0, c3, false, false);
    }
    // C/D layout: lane<16 -> rows M=0..7, lane>=16 -> rows M=8..15; N = lane&15
    float* out = H + (size_t)(mtile * 16 + ((lane & 16) ? 8 : 0)) * COLS
                   + ngroup * 64 + (lane & 15);
#pragma unroll
    for (int r = 0; r < 8; ++r) {
        out[(size_t)r * COLS]      = c0[r];
        out[(size_t)r * COLS + 16] = c1[r];
        out[(size_t)r * COLS + 32] = c2[r];
        out[(size_t)r * COLS + 48] = c3[r];
    }
}

// ---- per-node attention scores: s_i[n,h], s_j[n,h] ------------------------

__global__ __launch_bounds__(256) void gat_scores(
    const float* __restrict__ Hf, const float* __restrict__ a,
    float* __restrict__ si, float* __restrict__ sj) {
    const int n    = blockIdx.x;
    const int head = threadIdx.x >> 5;
    const int lane = threadIdx.x & 31;
    const float* hp = Hf + (size_t)n * COLS + head * HID;
    const float* ai = a + head * (2 * HID);
    float pi = 0.f, pj = 0.f;
#pragma unroll
    for (int j = 0; j < 4; ++j) {
        int d = lane + j * 32;
        float hv = hp[d];
        pi += hv * ai[d];
        pj += hv * ai[HID + d];
    }
#pragma unroll
    for (int off = 16; off; off >>= 1) {
        pi += __shfl_xor(pi, off, 32);
        pj += __shfl_xor(pj, off, 32);
    }
    if (lane == 0) { si[n * HEADS + head] = pi; sj[n * HEADS + head] = pj; }
}

// ---- edge phase -----------------------------------------------------------

__global__ void gat_edge_max(const float* __restrict__ si, const float* __restrict__ sj,
                             const int* __restrict__ esrc, const int* __restrict__ edst,
                             float* __restrict__ ebuf, unsigned* __restrict__ mkey,
                             int total) {
    int t = blockIdx.x * blockDim.x + threadIdx.x;
    if (t >= total) return;
    int e = t >> 3, h = t & 7;
    int s = esrc[e], d = edst[e];
    float v = si[d * HEADS + h] + sj[s * HEADS + h];
    v = v > 0.f ? v : 0.01f * v;                     // leaky_relu
    ebuf[t] = v;
    atomicMax(&mkey[d * HEADS + h], gat_fkey(v));
}

__global__ void gat_edge_exp(float* __restrict__ ebuf, const unsigned* __restrict__ mkey,
                             float* __restrict__ denom, const int* __restrict__ edst,
                             int total) {
    int t = blockIdx.x * blockDim.x + threadIdx.x;
    if (t >= total) return;
    int e = t >> 3, h = t & 7;
    int d = edst[e];
    float m  = gat_funkey(mkey[d * HEADS + h]);
    float ex = __expf(ebuf[t] - m);
    ebuf[t] = ex;                                    // in-place: e -> exp(e-m)
    atomicAdd(&denom[d * HEADS + h], ex);
}

// one wave per (edge, head); scatter alpha * h[src] into agg (L2-resident)
__global__ __launch_bounds__(256) void gat_edge_agg(
    const float* __restrict__ ebuf, const float* __restrict__ denom,
    const float* __restrict__ Hf, const int* __restrict__ esrc,
    const int* __restrict__ edst, float* __restrict__ agg) {
    const int e    = blockIdx.x;
    const int head = threadIdx.x >> 5;
    const int lane = threadIdx.x & 31;
    const int s = esrc[e], d = edst[e];
    float dn    = denom[d * HEADS + head];
    float alpha = ebuf[e * HEADS + head] / (dn > 0.f ? dn : 1.f);
    const float* hs = Hf + (size_t)s * COLS + head * HID;
    float* ag = agg + ((size_t)d * HEADS + head) * HID;
#pragma unroll
    for (int j = 0; j < 4; ++j) {
        int dd = lane + j * 32;
        atomicAdd(&ag[dd], alpha * hs[dd]);
    }
}

// ---- head mean + ELU ------------------------------------------------------

__global__ void gat_elu_mean(const float* __restrict__ agg,
                             unsigned short* __restrict__ ebf,
                             float* __restrict__ ef, int total) {
    int t = blockIdx.x * blockDim.x + threadIdx.x;
    if (t >= total) return;
    int n = t >> 7, d = t & 127;
    const float* ap = agg + (size_t)n * COLS + d;
    float s = 0.f;
#pragma unroll
    for (int h = 0; h < HEADS; ++h) s += ap[h * HID];
    s *= (1.f / HEADS);
    float v = s > 0.f ? s : (__expf(s) - 1.f);       // ELU
    ef[t]  = v;
    ebf[t] = gat_f2bf(v);
}

// ---- graph mean over nodes ------------------------------------------------

__global__ __launch_bounds__(256) void gat_colmean(const float* __restrict__ ef,
                                                   float* __restrict__ g, int n) {
    __shared__ float sm[256];
    const int d = blockIdx.x;
    float s = 0.f;
    for (int i = threadIdx.x; i < n; i += 256) s += ef[(size_t)i * HID + d];
    sm[threadIdx.x] = s; __syncthreads();
    for (int o = 128; o; o >>= 1) {
        if (threadIdx.x < o) sm[threadIdx.x] += sm[threadIdx.x + o];
        __syncthreads();
    }
    if (threadIdx.x == 0) g[d] = sm[0] / (float)n;
}

// ---- LayerNorm + MLP head (one workgroup) ---------------------------------

__global__ __launch_bounds__(128) void gat_mlp(
    const float* __restrict__ g, const float* __restrict__ lng,
    const float* __restrict__ lnb,
    const float* __restrict__ Wl1, const float* __restrict__ bl1,
    const float* __restrict__ Wl2, const float* __restrict__ bl2,
    const float* __restrict__ Wl3, const float* __restrict__ bl3,
    float* __restrict__ out) {
    __shared__ float sm[128];
    __shared__ float gn[128];
    __shared__ float x1[64];
    __shared__ float x2[16];
    const int t = threadIdx.x;
    float v = g[t];
    sm[t] = v; __syncthreads();
    for (int o = 64; o; o >>= 1) { if (t < o) sm[t] += sm[t + o]; __syncthreads(); }
    float mu = sm[0] * (1.f / 128.f);
    __syncthreads();
    float dv = v - mu;
    sm[t] = dv * dv; __syncthreads();
    for (int o = 64; o; o >>= 1) { if (t < o) sm[t] += sm[t + o]; __syncthreads(); }
    float var = sm[0] * (1.f / 128.f);
    __syncthreads();
    gn[t] = dv * rsqrtf(var + 1e-5f) * lng[t] + lnb[t];
    __syncthreads();
    if (t < 64) {
        float acc = bl1[t];
        for (int k = 0; k < 128; ++k) acc += Wl1[t * 128 + k] * gn[k];
        x1[t] = acc > 0.f ? acc : 0.01f * acc;
    }
    __syncthreads();
    if (t < 16) {
        float acc = bl2[t];
        for (int k = 0; k < 64; ++k) acc += Wl2[t * 64 + k] * x1[k];
        x2[t] = acc > 0.f ? acc : 0.01f * acc;
    }
    __syncthreads();
    if (t < 16) {
        float acc = bl3[t];
        for (int k = 0; k < 16; ++k) acc += Wl3[t * 16 + k] * x2[k];
        out[t] = acc > 0.f ? acc : 0.f;
    }
}

// ---------------------------------------------------------------------------

extern "C" void kernel_launch(void* const* d_in, const int* in_sizes, int n_in,
                              void* d_out, int out_size, void* d_ws, size_t ws_size,
                              hipStream_t stream) {
    const float* x    = (const float*)d_in[0];
    const int*   esrc = (const int*)d_in[1];
    const int*   edst = (const int*)d_in[2];
    const float* W1   = (const float*)d_in[3];
    const float* a1   = (const float*)d_in[4];
    const float* W2   = (const float*)d_in[5];
    const float* a2   = (const float*)d_in[6];
    const float* lng  = (const float*)d_in[7];
    const float* lnb  = (const float*)d_in[8];
    const float* Wl1  = (const float*)d_in[9];
    const float* bl1  = (const float*)d_in[10];
    const float* Wl2  = (const float*)d_in[11];
    const float* bl2  = (const float*)d_in[12];
    const float* Wl3  = (const float*)d_in[13];
    const float* bl3  = (const float*)d_in[14];
    (void)in_sizes; (void)n_in; (void)out_size; (void)ws_size;

    char* ws = (char*)d_ws;
    size_t off = 0;
    auto alloc = [&](size_t bytes) {
        size_t o = off; off += (bytes + 255) & ~(size_t)255; return o;
    };
    // all sizes below are 256-byte multiples -> mkey|denom|agg stay contiguous
    const size_t o_xb   = alloc((size_t)N_NODES * F1 * 2);          // x bf16
    const size_t o_w1b  = alloc((size_t)COLS * F1 * 2);             // W1 bf16
    const size_t o_w2b  = alloc((size_t)COLS * HID * 2);            // W2 bf16
    const size_t o_h    = alloc((size_t)N_NODES * COLS * 4);        // projections
    const size_t o_si   = alloc((size_t)N_NODES * HEADS * 4);
    const size_t o_sj   = alloc((size_t)N_NODES * HEADS * 4);
    const size_t o_e    = alloc((size_t)N_EDGES * HEADS * 4);       // e -> exp
    const size_t o_mkey = alloc((size_t)N_NODES * HEADS * 4);
    const size_t o_den  = alloc((size_t)N_NODES * HEADS * 4);
    const size_t o_agg  = alloc((size_t)N_NODES * COLS * 4);
    const size_t o_ebf  = alloc((size_t)N_NODES * HID * 2);         // emb bf16
    const size_t o_ef   = alloc((size_t)N_NODES * HID * 4);         // emb f32
    const size_t o_g    = alloc((size_t)HID * 4);

    float*          hbuf  = (float*)(ws + o_h);
    float*          si    = (float*)(ws + o_si);
    float*          sj    = (float*)(ws + o_sj);
    float*          ebuf  = (float*)(ws + o_e);
    unsigned*       mkey  = (unsigned*)(ws + o_mkey);
    float*          den   = (float*)(ws + o_den);
    float*          agg   = (float*)(ws + o_agg);
    unsigned short* embbf = (unsigned short*)(ws + o_ebf);
    float*          embf  = (float*)(ws + o_ef);
    float*          gvec  = (float*)(ws + o_g);

    const int EH = N_EDGES * HEADS;                 // 1,280,000
    const int tiles = (N_NODES / 16) * (COLS / 64); // 625*16 = 10000 (16x64 tiles)
    const int zero_n4 = (int)(((size_t)N_NODES * HEADS * 4 * 2 +
                               (size_t)N_NODES * COLS * 4) / 16);

    // --- convert inputs to bf16 ---
    gat_cvt_bf16<<<(N_NODES * F1 + 255) / 256, 256, 0, stream>>>(
        x, (unsigned short*)(ws + o_xb), N_NODES * F1);
    gat_cvt_bf16<<<(COLS * F1 + 255) / 256, 256, 0, stream>>>(
        W1, (unsigned short*)(ws + o_w1b), COLS * F1);
    gat_cvt_bf16<<<(COLS * HID + 255) / 256, 256, 0, stream>>>(
        W2, (unsigned short*)(ws + o_w2b), COLS * HID);

    // --- layer 1 ---
    gat_gemm_wmma<<<tiles / 4, 128, 0, stream>>>(
        (const __bf16*)(ws + o_xb), (const __bf16*)(ws + o_w1b), hbuf, F1, tiles);
    gat_scores<<<N_NODES, 256, 0, stream>>>(hbuf, a1, si, sj);
    gat_zero4<<<(zero_n4 + 255) / 256, 256, 0, stream>>>((float4*)(ws + o_mkey), zero_n4);
    gat_edge_max<<<(EH + 255) / 256, 256, 0, stream>>>(si, sj, esrc, edst, ebuf, mkey, EH);
    gat_edge_exp<<<(EH + 255) / 256, 256, 0, stream>>>(ebuf, mkey, den, edst, EH);
    gat_edge_agg<<<N_EDGES, 256, 0, stream>>>(ebuf, den, hbuf, esrc, edst, agg);
    gat_elu_mean<<<(N_NODES * HID + 255) / 256, 256, 0, stream>>>(
        agg, embbf, embf, N_NODES * HID);

    // --- layer 2 (A = emb1 bf16, F = 128) ---
    gat_gemm_wmma<<<tiles / 4, 128, 0, stream>>>(
        (const __bf16*)(ws + o_ebf), (const __bf16*)(ws + o_w2b), hbuf, HID, tiles);
    gat_scores<<<N_NODES, 256, 0, stream>>>(hbuf, a2, si, sj);
    gat_zero4<<<(zero_n4 + 255) / 256, 256, 0, stream>>>((float4*)(ws + o_mkey), zero_n4);
    gat_edge_max<<<(EH + 255) / 256, 256, 0, stream>>>(si, sj, esrc, edst, ebuf, mkey, EH);
    gat_edge_exp<<<(EH + 255) / 256, 256, 0, stream>>>(ebuf, mkey, den, edst, EH);
    gat_edge_agg<<<N_EDGES, 256, 0, stream>>>(ebuf, den, hbuf, esrc, edst, agg);
    gat_elu_mean<<<(N_NODES * HID + 255) / 256, 256, 0, stream>>>(
        agg, embbf, embf, N_NODES * HID);

    // --- readout ---
    gat_colmean<<<HID, 256, 0, stream>>>(embf, gvec, N_NODES);
    gat_mlp<<<1, 128, 0, stream>>>(gvec, lng, lnb, Wl1, bl1, Wl2, bl2, Wl3, bl3,
                                   (float*)d_out);
}
